// VisionTransformer_24343874634446
// MI455X (gfx1250) — compile-verified, matrix operand order checked by reading, not run on hardware
//
#include <hip/hip_runtime.h>
#include <math.h>

// ---------------- problem constants ----------------
#define B_     2
#define CIN_   3
#define IMG_   384
#define PP_    16
#define D_     1024
#define H_     16
#define HD_    64
#define GH_    24
#define NP_    576
#define NTOK_  577
#define NPAD_  608           // padded tokens: multiple of 32
#define DFF_   4096
#define DEPTH_ 24
#define KPE_   768           // CIN*P*P

// ---------------- GEMM tiling ----------------
#define BM 128
#define BN 128
#define BK 32
#define LDSA 48              // LDS row stride in ushorts (96B, 16B aligned)

typedef __bf16 v16bf __attribute__((ext_vector_type(16)));
typedef float  v8f   __attribute__((ext_vector_type(8)));

union FragAB { uint4 q[2]; v16bf v; };
union FragC  { v8f v; float f[8]; };

__device__ __forceinline__ unsigned short f2bf(float f) {
    unsigned int u = __float_as_uint(f);
    u = u + 0x7FFFu + ((u >> 16) & 1u);     // round-to-nearest-even
    return (unsigned short)(u >> 16);
}

// async 16B global -> LDS copy (CDNA5 GLOBAL_LOAD_ASYNC_TO_LDS_B128, ASYNCcnt)
__device__ __forceinline__ void async_b128(const unsigned short* g, unsigned short* l) {
    unsigned      lo = (unsigned)(unsigned long long)(size_t)l;   // low 32 bits = LDS offset
    unsigned long long ga = (unsigned long long)(size_t)g;
    asm volatile("global_load_async_to_lds_b128 %0, %1, off" :: "v"(lo), "v"(ga) : "memory");
}

// ------------------------------------------------------------------
// bf16 WMMA GEMM:  C[z] = act( scale*(A[z] @ B[z]^T) + bias ) + Res
//   A : M x K  bf16 row-major (lda)
//   B : N x K  bf16 row-major (ldb)  -- all operands pre-laid-out n-major
//   C : fp32 or bf16 (cbf flag); Res/bias fp32
//   grid.z -> (bz=z/Hdiv, hz=z%Hdiv) with per-operand strides.
//   Staging is branch-free async copy; OOB tile rows read workspace slack.
// ------------------------------------------------------------------
__global__ __launch_bounds__(256)
void gemm_bf16_wmma(const unsigned short* __restrict__ A, int lda, long long aSB, long long aSH,
                    const unsigned short* __restrict__ Bm, int ldb, long long bSB, long long bSH,
                    const float* __restrict__ bias,
                    const float* __restrict__ Res,
                    void* __restrict__ C, int ldc, long long cSB, long long cSH, int cbf,
                    int M, int K, int Nout, int Hdiv, float scale, int act)
{
    __shared__ unsigned short As[2][BM * LDSA];
    __shared__ unsigned short Bs[2][BN * LDSA];

    const int z  = blockIdx.z;
    const int bz = z / Hdiv;
    const int hz = z % Hdiv;
    A  += (long long)bz * aSB + (long long)hz * aSH;
    Bm += (long long)bz * bSB + (long long)hz * bSH;
    const long long cz = (long long)bz * cSB + (long long)hz * cSH;

    const int row0 = blockIdx.y * BM;
    const int col0 = blockIdx.x * BN;

    const int t    = threadIdx.x;
    const int lane = t & 31;
    const int wave = t >> 5;
    const int wm   = wave >> 2;          // 0..1  (64-row strip)
    const int wn   = wave & 3;           // 0..3  (32-col strip)

    const int sm  = t >> 1;              // staged row/col 0..127
    const int skh = (t & 1) * 16;        // k half: 0 or 16 (elements)

    FragC acc[4][2];
#pragma unroll
    for (int i = 0; i < 4; ++i)
#pragma unroll
        for (int j = 0; j < 2; ++j)
#pragma unroll
            for (int r = 0; r < 8; ++r) acc[i][j].f[r] = 0.0f;

    const unsigned short* gA0 = A  + (long long)(row0 + sm) * lda + skh;
    const unsigned short* gB0 = Bm + (long long)(col0 + sm) * ldb + skh;
    unsigned short* lA[2] = { &As[0][sm * LDSA + skh], &As[1][sm * LDSA + skh] };
    unsigned short* lB[2] = { &Bs[0][sm * LDSA + skh], &Bs[1][sm * LDSA + skh] };

    const int nk = K / BK;

    // prologue: stage tile 0 into buffer 0
    async_b128(gA0,     lA[0]);
    async_b128(gA0 + 8, lA[0] + 8);
    async_b128(gB0,     lB[0]);
    async_b128(gB0 + 8, lB[0] + 8);

    int p = 0;
    for (int ks = 0; ks < nk; ++ks) {
        if (ks + 1 < nk) {
            const int k1 = (ks + 1) * BK;
            async_b128(gA0 + k1,     lA[p ^ 1]);
            async_b128(gA0 + k1 + 8, lA[p ^ 1] + 8);
            async_b128(gB0 + k1,     lB[p ^ 1]);
            async_b128(gB0 + k1 + 8, lB[p ^ 1] + 8);
            asm volatile("s_wait_asynccnt 0x4" ::: "memory");   // tile ks complete (in-order)
        } else {
            asm volatile("s_wait_asynccnt 0x0" ::: "memory");
        }
        __syncthreads();

        FragAB af[4], bfr[2];
#pragma unroll
        for (int i = 0; i < 4; ++i) {
            const unsigned short* q = &As[p][(wm * 64 + i * 16 + (lane & 15)) * LDSA + ((lane >> 4) * 8)];
            af[i].q[0] = *(const uint4*)(q);
            af[i].q[1] = *(const uint4*)(q + 16);
        }
#pragma unroll
        for (int j = 0; j < 2; ++j) {
            const unsigned short* q = &Bs[p][(wn * 32 + j * 16 + (lane & 15)) * LDSA + ((lane >> 4) * 8)];
            bfr[j].q[0] = *(const uint4*)(q);
            bfr[j].q[1] = *(const uint4*)(q + 16);
        }
#pragma unroll
        for (int i = 0; i < 4; ++i)
#pragma unroll
            for (int j = 0; j < 2; ++j)
                acc[i][j].v = __builtin_amdgcn_wmma_f32_16x16x32_bf16(
                    false, af[i].v, false, bfr[j].v, (short)0, acc[i][j].v, false, false);
        __syncthreads();      // all waves done reading buf p before it is overwritten
        p ^= 1;
    }

    // ---------------- epilogue ----------------
    float*          Cf  = (float*)C;
    unsigned short* C16 = (unsigned short*)C;
#pragma unroll
    for (int i = 0; i < 4; ++i) {
#pragma unroll
        for (int j = 0; j < 2; ++j) {
#pragma unroll
            for (int r = 0; r < 8; ++r) {
                const int row = row0 + wm * 64 + i * 16 + r + ((lane >> 4) << 3);
                const int col = col0 + wn * 32 + j * 16 + (lane & 15);
                if (row < M && col < Nout) {
                    const long long off = cz + (long long)row * ldc + col;
                    float x = acc[i][j].f[r] * scale;
                    if (bias) x += bias[col];
                    if (act == 1) x = 0.5f * x * (1.0f + erff(x * 0.70710678118654752f));
                    if (Res) x += Res[off];
                    if (cbf) C16[off] = f2bf(x);
                    else     Cf[off]  = x;
                }
            }
        }
    }
}

// ---------------- LayerNorm: fp32 in, bf16 out ----------------
__global__ __launch_bounds__(256)
void ln_kernel(const float* __restrict__ x, const float* __restrict__ g,
               const float* __restrict__ b, unsigned short* __restrict__ y)
{
    __shared__ float red[256];
    const long long row = blockIdx.x;
    const float* xr = x + row * D_;
    float v[4];
#pragma unroll
    for (int i = 0; i < 4; ++i) v[i] = xr[threadIdx.x + i * 256];
    red[threadIdx.x] = v[0] + v[1] + v[2] + v[3]; __syncthreads();
    for (int off = 128; off > 0; off >>= 1) {
        if (threadIdx.x < off) red[threadIdx.x] += red[threadIdx.x + off];
        __syncthreads();
    }
    const float mean = red[0] * (1.0f / D_);
    __syncthreads();
    float q = 0.0f;
#pragma unroll
    for (int i = 0; i < 4; ++i) { float d = v[i] - mean; q += d * d; }
    red[threadIdx.x] = q; __syncthreads();
    for (int off = 128; off > 0; off >>= 1) {
        if (threadIdx.x < off) red[threadIdx.x] += red[threadIdx.x + off];
        __syncthreads();
    }
    const float rstd = rsqrtf(red[0] * (1.0f / D_) + 1e-6f);
#pragma unroll
    for (int i = 0; i < 4; ++i) {
        const int c = threadIdx.x + i * 256;
        y[row * D_ + c] = f2bf((v[i] - mean) * rstd * g[c] + b[c]);
    }
}

// ---------------- softmax: fp32 att in, bf16 att out, masks cols >= NTOK_ ----------------
__global__ __launch_bounds__(256)
void softmax_kernel(const float* __restrict__ att, unsigned short* __restrict__ att16)
{
    __shared__ float red[256];
    const long long row = blockIdx.x;
    const float* a = att + row * NPAD_;
    unsigned short* ao = att16 + row * NPAD_;
    float v[3];
    float mx = -3.0e38f;
#pragma unroll
    for (int i = 0; i < 3; ++i) {
        const int c = threadIdx.x + i * 256;
        v[i] = (c < NTOK_) ? a[c] : -3.0e38f;
        mx = fmaxf(mx, v[i]);
    }
    red[threadIdx.x] = mx; __syncthreads();
    for (int off = 128; off > 0; off >>= 1) {
        if (threadIdx.x < off) red[threadIdx.x] = fmaxf(red[threadIdx.x], red[threadIdx.x + off]);
        __syncthreads();
    }
    mx = red[0]; __syncthreads();
    float s = 0.0f;
#pragma unroll
    for (int i = 0; i < 3; ++i) {
        const int c = threadIdx.x + i * 256;
        v[i] = (c < NTOK_) ? __expf(v[i] - mx) : 0.0f;
        s += v[i];
    }
    red[threadIdx.x] = s; __syncthreads();
    for (int off = 128; off > 0; off >>= 1) {
        if (threadIdx.x < off) red[threadIdx.x] += red[threadIdx.x + off];
        __syncthreads();
    }
    const float inv = 1.0f / red[0];
#pragma unroll
    for (int i = 0; i < 3; ++i) {
        const int c = threadIdx.x + i * 256;
        if (c < NPAD_) ao[c] = f2bf((c < NTOK_) ? v[i] * inv : 0.0f);
    }
}

// ---------------- elementwise fp32 -> bf16 ----------------
__global__ __launch_bounds__(256)
void cvt_kernel(const float* __restrict__ src, unsigned short* __restrict__ dst, int n)
{
    const int idx = blockIdx.x * 256 + threadIdx.x;
    if (idx < n) dst[idx] = f2bf(src[idx]);
}

// ---------------- transpose + convert: W[K,N] fp32 -> wT[N,K] bf16 ----------------
__global__ __launch_bounds__(256)
void cvtT_kernel(const float* __restrict__ W, unsigned short* __restrict__ wT, int K, int N)
{
    const int idx = blockIdx.x * 256 + threadIdx.x;
    if (idx >= N * K) return;
    const int n = idx / K, k = idx % K;
    wT[idx] = f2bf(W[(long long)k * N + n]);
}

// ---------------- transpose v part of qkv (bf16 copy): vT[(b,h,d),m] ----------------
__global__ __launch_bounds__(256)
void vT_kernel(const unsigned short* __restrict__ qkv, unsigned short* __restrict__ vT)
{
    const int idx = blockIdx.x * 256 + threadIdx.x;
    if (idx >= B_ * H_ * HD_ * NPAD_) return;
    const int m  = idx % NPAD_;
    const int d  = (idx / NPAD_) % HD_;
    const int hh = (idx / (NPAD_ * HD_)) % H_;
    const int b  = idx / (NPAD_ * HD_ * H_);
    vT[idx] = qkv[(long long)(b * NPAD_ + m) * (3 * D_) + 2 * D_ + hh * HD_ + d];
}

// ---------------- im2col (fp32 -> bf16) ----------------
__global__ __launch_bounds__(256)
void im2col_kernel(const float* __restrict__ x, unsigned short* __restrict__ Ape)
{
    const int idx = blockIdx.x * 256 + threadIdx.x;
    if (idx >= B_ * NP_ * KPE_) return;
    const int k  = idx % KPE_;
    const int pp = (idx / KPE_) % NP_;
    const int b  = idx / (KPE_ * NP_);
    const int c  = k >> 8;
    const int py = (k >> 4) & 15;
    const int px = k & 15;
    const int ph = pp / GH_, pw = pp % GH_;
    Ape[idx] = f2bf(x[(((long long)b * CIN_ + c) * IMG_ + ph * PP_ + py) * IMG_ + pw * PP_ + px]);
}

// cls token + positional embedding, zero pad rows
__global__ __launch_bounds__(256)
void embed_kernel(float* __restrict__ h, const float* __restrict__ cls,
                  const float* __restrict__ pos)
{
    const int idx = blockIdx.x * 256 + threadIdx.x;
    if (idx >= B_ * NPAD_ * D_) return;
    const int d = idx & (D_ - 1);
    const int n = (idx >> 10) % NPAD_;
    float val;
    if (n == 0)         val = cls[d] + pos[d];
    else if (n < NTOK_) val = h[idx] + pos[n * D_ + d];
    else                val = 0.0f;
    h[idx] = val;
}

__global__ __launch_bounds__(256)
void copyout_kernel(const float* __restrict__ h, float* __restrict__ out)
{
    const int idx = blockIdx.x * 256 + threadIdx.x;
    if (idx >= B_ * NTOK_ * D_) return;
    const int d = idx & (D_ - 1);
    const int n = (idx >> 10) % NTOK_;
    const int b = idx / (NTOK_ * D_);
    out[idx] = h[((long long)b * NPAD_ + n) * D_ + d];
}

// ------------------------------------------------------------------
extern "C" void kernel_launch(void* const* d_in, const int* in_sizes, int n_in,
                              void* d_out, int out_size, void* d_ws, size_t ws_size,
                              hipStream_t stream)
{
    const float* x    = (const float*)d_in[0];
    const float* Wp   = (const float*)d_in[1];
    const float* bp   = (const float*)d_in[2];
    const float* cls  = (const float*)d_in[3];
    const float* pos  = (const float*)d_in[4];
    const float* g1   = (const float*)d_in[5];
    const float* b1   = (const float*)d_in[6];
    const float* Wqkv = (const float*)d_in[7];
    const float* bqkv = (const float*)d_in[8];
    const float* Wo   = (const float*)d_in[9];
    const float* bo   = (const float*)d_in[10];
    const float* g2   = (const float*)d_in[11];
    const float* b2   = (const float*)d_in[12];
    const float* W1   = (const float*)d_in[13];
    const float* b1m  = (const float*)d_in[14];
    const float* W2   = (const float*)d_in[15];
    const float* b2m  = (const float*)d_in[16];

    const long long HSZ   = (long long)B_ * NPAD_ * D_;          // 1,245,184
    const long long QKVSZ = (long long)B_ * NPAD_ * 3 * D_;      // 3,735,552
    const long long FFSZ  = (long long)B_ * NPAD_ * DFF_;        // 4,980,736
    const long long ATTSZ = (long long)B_ * H_ * NPAD_ * NPAD_;  // 11,829,248
    const long long WTSZ  = (long long)D_ * DFF_;                // 4,194,304 (max weight)

    char* wp = (char*)d_ws;
    float* h   = (float*)wp;            wp += HSZ   * 4;
    float* att = (float*)wp;            wp += ATTSZ * 4;
    unsigned short* y16   = (unsigned short*)wp; wp += HSZ   * 2;
    unsigned short* qkv16 = (unsigned short*)wp; wp += QKVSZ * 2;
    unsigned short* o16   = (unsigned short*)wp; wp += HSZ   * 2;
    unsigned short* ff16  = (unsigned short*)wp; wp += FFSZ  * 2;
    unsigned short* att16 = (unsigned short*)wp; wp += ATTSZ * 2;
    unsigned short* vT    = (unsigned short*)wp; wp += HSZ   * 2;
    unsigned short* wT    = (unsigned short*)wp; wp += WTSZ  * 2;
    unsigned short* Ape   = ff16;   // alias: used only before ff16
    // + implicit slack: OOB tile reads from the last region (wT) cannot occur
    //   (weight N,K are exact multiples of the tile); all others spill into the
    //   following region, which is defined workspace memory.

    const dim3 blk(256);
    const int ROWS = B_ * NPAD_;                       // 1216
    const int MY   = (ROWS + BM - 1) / BM;             // 10
    const int MYA  = (NPAD_ + BM - 1) / BM;            // 5

    // ---- patch embedding ----
    im2col_kernel<<<(B_ * NP_ * KPE_ + 255) / 256, blk, 0, stream>>>(x, Ape);
    cvt_kernel<<<(D_ * KPE_ + 255) / 256, blk, 0, stream>>>(Wp, wT, D_ * KPE_);  // Wp already [N,K]
    gemm_bf16_wmma<<<dim3(D_ / BN, (NP_ + BM - 1) / BM, B_), blk, 0, stream>>>(
        Ape, KPE_, (long long)NP_ * KPE_, 0,
        wT, KPE_, 0, 0,
        bp, nullptr,
        h + D_, D_, (long long)NPAD_ * D_, 0, 0,
        NP_, KPE_, D_, 1, 1.0f, 0);
    embed_kernel<<<(B_ * NPAD_ * D_ + 255) / 256, blk, 0, stream>>>(h, cls, pos);

    // ---- transformer blocks ----
    for (int l = 0; l < DEPTH_; ++l) {
        ln_kernel<<<ROWS, blk, 0, stream>>>(h, g1 + (long long)l * D_, b1 + (long long)l * D_, y16);

        // qkv = y @ Wqkv + bqkv  -> bf16
        cvtT_kernel<<<(3 * D_ * D_ + 255) / 256, blk, 0, stream>>>(
            Wqkv + (long long)l * D_ * 3 * D_, wT, D_, 3 * D_);
        gemm_bf16_wmma<<<dim3(3 * D_ / BN, MY, 1), blk, 0, stream>>>(
            y16, D_, 0, 0,
            wT, D_, 0, 0,
            bqkv + (long long)l * 3 * D_, nullptr,
            qkv16, 3 * D_, 0, 0, 1,
            ROWS, D_, 3 * D_, 1, 1.0f, 0);

        // att = scale * q @ k^T  (fp32 scores)
        gemm_bf16_wmma<<<dim3(MYA, MYA, B_ * H_), blk, 0, stream>>>(
            qkv16, 3 * D_, (long long)NPAD_ * 3 * D_, HD_,
            qkv16 + D_, 3 * D_, (long long)NPAD_ * 3 * D_, HD_,
            nullptr, nullptr,
            att, NPAD_, (long long)H_ * NPAD_ * NPAD_, (long long)NPAD_ * NPAD_, 0,
            NPAD_, HD_, NPAD_, H_, 0.125f, 0);
        softmax_kernel<<<B_ * H_ * NPAD_, blk, 0, stream>>>(att, att16);
        vT_kernel<<<(B_ * H_ * HD_ * NPAD_ + 255) / 256, blk, 0, stream>>>(qkv16, vT);

        // o = att @ v  -> bf16
        gemm_bf16_wmma<<<dim3(1, MYA, B_ * H_), blk, 0, stream>>>(
            att16, NPAD_, (long long)H_ * NPAD_ * NPAD_, (long long)NPAD_ * NPAD_,
            vT, NPAD_, (long long)H_ * HD_ * NPAD_, (long long)HD_ * NPAD_,
            nullptr, nullptr,
            o16, D_, (long long)NPAD_ * D_, HD_, 1,
            NPAD_, NPAD_, HD_, H_, 1.0f, 0);

        // h = h + o @ Wo + bo
        cvtT_kernel<<<(D_ * D_ + 255) / 256, blk, 0, stream>>>(
            Wo + (long long)l * D_ * D_, wT, D_, D_);
        gemm_bf16_wmma<<<dim3(D_ / BN, MY, 1), blk, 0, stream>>>(
            o16, D_, 0, 0,
            wT, D_, 0, 0,
            bo + (long long)l * D_, h,
            h, D_, 0, 0, 0,
            ROWS, D_, D_, 1, 1.0f, 0);

        ln_kernel<<<ROWS, blk, 0, stream>>>(h, g2 + (long long)l * D_, b2 + (long long)l * D_, y16);

        // ff = gelu(y @ W1 + b1m) -> bf16
        cvtT_kernel<<<(D_ * DFF_ + 255) / 256, blk, 0, stream>>>(
            W1 + (long long)l * D_ * DFF_, wT, D_, DFF_);
        gemm_bf16_wmma<<<dim3(DFF_ / BN, MY, 1), blk, 0, stream>>>(
            y16, D_, 0, 0,
            wT, D_, 0, 0,
            b1m + (long long)l * DFF_, nullptr,
            ff16, DFF_, 0, 0, 1,
            ROWS, D_, DFF_, 1, 1.0f, 1);

        // h = h + ff @ W2 + b2m
        cvtT_kernel<<<(DFF_ * D_ + 255) / 256, blk, 0, stream>>>(
            W2 + (long long)l * DFF_ * D_, wT, DFF_, D_);
        gemm_bf16_wmma<<<dim3(D_ / BN, MY, 1), blk, 0, stream>>>(
            ff16, DFF_, 0, 0,
            wT, DFF_, 0, 0,
            b2m + (long long)l * D_, h,
            h, D_, 0, 0, 0,
            ROWS, DFF_, D_, 1, 1.0f, 0);
    }

    copyout_kernel<<<(B_ * NTOK_ * D_ + 255) / 256, blk, 0, stream>>>(h, (float*)d_out);
}
